// AttentionFlow_34969623724767
// MI455X (gfx1250) — matrix operand order repaired and checked by reading, not compiled, not run on hardware
//
#include <hip/hip_runtime.h>
#include <hip/hip_bf16.h>

typedef __attribute__((ext_vector_type(16))) __bf16 v16bf;
typedef __attribute__((ext_vector_type(8)))  float  v8f;

union BF16x16 { v16bf v; uint4 q[2]; };

// ---------- helpers ----------

__device__ __forceinline__ unsigned f2ord(float f) {
    unsigned u = __float_as_uint(f);
    return (u & 0x80000000u) ? ~u : (u | 0x80000000u);
}
__device__ __forceinline__ float ord2f(unsigned u) {
    return __uint_as_float((u & 0x80000000u) ? (u ^ 0x80000000u) : ~u);
}

__device__ __forceinline__ v8f wmma_bf16(v16bf a, v16bf b, v8f c) {
    return __builtin_amdgcn_wmma_f32_16x16x32_bf16(false, a, false, b, (short)0, c,
                                                   false, false);
}

// A-fragment (16x32 bf16) from a row-major fp32 row.
// lane<16 holds K {0..7, 16..23} (koff=0); lane>=16 holds K {8..15, 24..31} (koff=8).
__device__ __forceinline__ v16bf a_frag_f32(const float* __restrict__ rowp, int kc, int koff) {
    const float4* q = (const float4*)(rowp + kc * 32 + koff);
    float4 x0 = q[0], x1 = q[1], x2 = q[4], x3 = q[5];
    v16bf a;
    a[0]  = (__bf16)x0.x; a[1]  = (__bf16)x0.y; a[2]  = (__bf16)x0.z; a[3]  = (__bf16)x0.w;
    a[4]  = (__bf16)x1.x; a[5]  = (__bf16)x1.y; a[6]  = (__bf16)x1.z; a[7]  = (__bf16)x1.w;
    a[8]  = (__bf16)x2.x; a[9]  = (__bf16)x2.y; a[10] = (__bf16)x2.z; a[11] = (__bf16)x2.w;
    a[12] = (__bf16)x3.x; a[13] = (__bf16)x3.y; a[14] = (__bf16)x3.z; a[15] = (__bf16)x3.w;
    return a;
}

// A-fragment from a row-major bf16 row (row length 64, used for the center GEMM).
__device__ __forceinline__ v16bf a_frag_bf(const __bf16* rowp, int kc, int koff) {
    const __bf16* p = rowp + kc * 32 + koff;
    BF16x16 t;
    t.q[0] = *(const uint4*)(p);
    t.q[1] = *(const uint4*)(p + 16);
    return t.v;
}

// B-fragment (32x16 bf16) from row-major W[n][k] (out x in). Lane holds K
// {0..7, 16..23} (lane<16) or {8..15, 24..31} (lane>=16) of column n = nt*16 + (lane&15).
__device__ __forceinline__ v16bf b_frag(const __bf16* base, int rowK, int nt, int kc, int lane) {
    const __bf16* p = base + (size_t)(nt * 16 + (lane & 15)) * rowK + kc * 32 +
                      ((lane < 16) ? 0 : 8);
    BF16x16 t;
    t.q[0] = *(const uint4*)(p);
    t.q[1] = *(const uint4*)(p + 16);
    return t.v;
}

// ---------- kernel 1: init ----------

__global__ void init_kernel(float* out, unsigned* segmax, float* segsum, int N, int outN) {
    int i = blockIdx.x * blockDim.x + threadIdx.x;
    if (i < outN) out[i] = 0.f;
    if (i < N) { segmax[i] = 0u; segsum[i] = 0.f; }
}

// ---------- kernel 2: prep (fused weights + per-batch bias) ----------

__global__ __launch_bounds__(256) void prep_kernel(
    const float* __restrict__ qse, const float* __restrict__ qre, const float* __restrict__ qte,
    const float* __restrict__ Wproj, const float* __restrict__ bproj,
    const float* __restrict__ Wst, const float* __restrict__ bst,
    const float* __restrict__ Wtm, const float* __restrict__ btm,
    const float* __restrict__ Wl, const float* __restrict__ bl,
    const float* __restrict__ Wr, const float* __restrict__ br,
    const float* __restrict__ Wcf,
    float* qbiasL, float* qbiasR,
    __bf16* WLi, __bf16* WLrel, __bf16* WRj, __bf16* WRrel, __bf16* WcO, int B)
{
    __shared__ float qcat[64 * 192];
    int tid = threadIdx.x;
    // q_cat[b][g]
    for (int idx = tid; idx < B * 192; idx += 256) {
        int bb = idx / 192, g = idx % 192;
        const float* w; const float* x; float bias;
        if (g < 64)       { w = Wst   + g * 128;        x = qse + bb * 128; bias = bst[g]; }
        else if (g < 128) { w = Wproj + (g - 64) * 128; x = qre + bb * 128; bias = bproj[g - 64]; }
        else              { w = Wtm   + (g - 128) * 128; x = qte + bb * 128; bias = btm[g - 128]; }
        float s = bias;
        for (int c = 0; c < 128; ++c) s += w[c] * x[c];
        qcat[idx] = s;
    }
    __syncthreads();
    // qbiasL/R[b][o] = b + W[:,128:320]@qcat + (W[:,0:64]+W[:,64:128])@b_proj
    for (int idx = tid; idx < B * 64; idx += 256) {
        int bb = idx / 64, o = idx % 64;
        float sl = bl[o], sr = br[o];
        const float* ql = Wl + o * 320 + 128;
        const float* qr = Wr + o * 320 + 128;
        const float* qc = qcat + bb * 192;
        for (int g = 0; g < 192; ++g) { sl += ql[g] * qc[g]; sr += qr[g] * qc[g]; }
        for (int c = 0; c < 64; ++c) {
            float bp = bproj[c];
            sl += (Wl[o * 320 + c] + Wl[o * 320 + 64 + c]) * bp;
            sr += (Wr[o * 320 + c] + Wr[o * 320 + 64 + c]) * bp;
        }
        qbiasL[idx] = sl; qbiasR[idx] = sr;
    }
    // Fused 64x128 weights (bf16, row-major out x in)
    for (int idx = tid; idx < 64 * 128; idx += 256) {
        int o = idx / 128, kk = idx % 128;
        float s0 = 0.f, s1 = 0.f, s2 = 0.f, s3 = 0.f;
        for (int c = 0; c < 64; ++c) {
            float wp = Wproj[c * 128 + kk];
            s0 += Wl[o * 320 + c]      * wp;
            s1 += Wl[o * 320 + 64 + c] * wp;
            s2 += Wr[o * 320 + c]      * wp;
            s3 += Wr[o * 320 + 64 + c] * wp;
        }
        WLi[idx]   = (__bf16)s0; WLrel[idx] = (__bf16)s1;
        WRj[idx]   = (__bf16)s2; WRrel[idx] = (__bf16)s3;
    }
    for (int idx = tid; idx < 64 * 64; idx += 256) WcO[idx] = (__bf16)Wcf[idx];
}

// ---------- kernel 3: per-edge WMMA pipeline -> logits + segment max ----------

__global__ __launch_bounds__(256) void edge_wmma_kernel(
    const int* __restrict__ edges,
    const float* __restrict__ emb,
    const float* __restrict__ rel,
    const float* __restrict__ qbiasL,
    const float* __restrict__ qbiasR,
    const __bf16* __restrict__ gWLi, const __bf16* __restrict__ gWLrel,
    const __bf16* __restrict__ gWRj, const __bf16* __restrict__ gWRrel,
    const __bf16* __restrict__ gWc,
    const float* __restrict__ bcenter,
    float* __restrict__ logits,
    unsigned* __restrict__ segmax)
{
    extern __shared__ __attribute__((aligned(16))) char smem[];
    __bf16* sWLi   = (__bf16*)(smem);
    __bf16* sWLrel = (__bf16*)(smem + 16384);
    __bf16* sWRj   = (__bf16*)(smem + 32768);
    __bf16* sWRrel = (__bf16*)(smem + 49152);
    __bf16* sWc    = (__bf16*)(smem + 65536);
    __bf16* sR     = (__bf16*)(smem + 73728);   // 8 waves * 16*64 bf16

    int tid = threadIdx.x;
    {   // stage weights to LDS (uint4 copies)
        uint4*       d[4] = { (uint4*)sWLi, (uint4*)sWLrel, (uint4*)sWRj, (uint4*)sWRrel };
        const uint4* s[4] = { (const uint4*)gWLi, (const uint4*)gWLrel,
                              (const uint4*)gWRj, (const uint4*)gWRrel };
        for (int a = 0; a < 4; ++a)
            for (int i = tid; i < 1024; i += 256) d[a][i] = s[a][i];
        uint4* dc = (uint4*)sWc; const uint4* sc = (const uint4*)gWc;
        for (int i = tid; i < 512; i += 256) dc[i] = sc[i];
    }
    __syncthreads();

    int wave = tid >> 5;
    int lane = tid & 31;
    long tile = (long)blockIdx.x * 8 + wave;
    long e0 = tile * 16;                 // 16 edges per tile; 4096 % 16 == 0
    int m    = lane & 15;
    int half = (lane < 16) ? 0 : 1;
    int koff = half * 8;
    int ncol = lane & 15;
    long em = e0 + m;

    int bb = edges[e0 * 8 + 0];
    int ii = edges[em * 8 + 6];
    int jj = edges[em * 8 + 7];
    const float* rowI = emb + (size_t)ii * 128;
    const float* rowJ = emb + (size_t)jj * 128;
    const float* rowR = rel + (size_t)em * 128;

    v8f cl[4], cr[4];
    #pragma unroll
    for (int nt = 0; nt < 4; ++nt)
        #pragma unroll
        for (int v = 0; v < 8; ++v) { cl[nt][v] = 0.f; cr[nt][v] = 0.f; }

    #pragma unroll
    for (int kc = 0; kc < 4; ++kc) {
        v16bf ai = a_frag_f32(rowI, kc, koff);
        v16bf aj = a_frag_f32(rowJ, kc, koff);
        v16bf ar = a_frag_f32(rowR, kc, koff);
        #pragma unroll
        for (int nt = 0; nt < 4; ++nt) {
            cl[nt] = wmma_bf16(ai, b_frag(sWLi,   128, nt, kc, lane), cl[nt]);
            cl[nt] = wmma_bf16(ar, b_frag(sWLrel, 128, nt, kc, lane), cl[nt]);
            cr[nt] = wmma_bf16(aj, b_frag(sWRj,   128, nt, kc, lane), cr[nt]);
            cr[nt] = wmma_bf16(ar, b_frag(sWRrel, 128, nt, kc, lane), cr[nt]);
        }
    }

    // bias + leaky; keep l in regs, spill leaky(r_pre) to LDS as bf16 for center GEMM
    __bf16* myR = sR + wave * 1024;
    #pragma unroll
    for (int nt = 0; nt < 4; ++nt) {
        float ql = qbiasL[bb * 64 + nt * 16 + ncol];
        float qr = qbiasR[bb * 64 + nt * 16 + ncol];
        #pragma unroll
        for (int v = 0; v < 8; ++v) {
            float xl = cl[nt][v] + ql;
            cl[nt][v] = (xl >= 0.f) ? xl : 0.01f * xl;
            float xr = cr[nt][v] + qr;
            xr = (xr >= 0.f) ? xr : 0.01f * xr;
            myR[(v + half * 8) * 64 + nt * 16 + ncol] = (__bf16)xr;
        }
    }
    __syncthreads();

    // center: (16x64) @ W_center^T -> 16x64
    v8f cc[4];
    #pragma unroll
    for (int nt = 0; nt < 4; ++nt)
        #pragma unroll
        for (int v = 0; v < 8; ++v) cc[nt][v] = 0.f;
    #pragma unroll
    for (int kc = 0; kc < 2; ++kc) {
        v16bf a = a_frag_bf(myR + m * 64, kc, koff);
        #pragma unroll
        for (int nt = 0; nt < 4; ++nt)
            cc[nt] = wmma_bf16(a, b_frag(sWc, 64, nt, kc, lane), cc[nt]);
    }

    // logit = sum_n l[m][n] * (center[m][n] + b_center[n]); reduce across 16 lanes
    float s[8];
    #pragma unroll
    for (int v = 0; v < 8; ++v) s[v] = 0.f;
    #pragma unroll
    for (int nt = 0; nt < 4; ++nt) {
        float bcn = bcenter[nt * 16 + ncol];
        #pragma unroll
        for (int v = 0; v < 8; ++v) s[v] += cl[nt][v] * (cc[nt][v] + bcn);
    }
    #pragma unroll
    for (int v = 0; v < 8; ++v) {
        s[v] += __shfl_xor(s[v], 1, 32);
        s[v] += __shfl_xor(s[v], 2, 32);
        s[v] += __shfl_xor(s[v], 4, 32);
        s[v] += __shfl_xor(s[v], 8, 32);
    }
    if (ncol == 0) {
        int base = half * 8;
        #pragma unroll
        for (int v = 0; v < 8; ++v) {
            long e = e0 + base + v;
            logits[e] = s[v];
            int node = edges[e * 8 + 6];
            atomicMax(segmax + node, f2ord(s[v]));
        }
    }
}

// ---------- kernel 4: exp + segment sum ----------

__global__ void softmax_sum_kernel(const int* __restrict__ edges,
                                   const float* __restrict__ logits,
                                   const unsigned* __restrict__ segmax,
                                   float* __restrict__ exbuf,
                                   float* __restrict__ segsum, long E)
{
    long e = (long)blockIdx.x * blockDim.x + threadIdx.x;
    if (e >= E) return;
    int ii = edges[e * 8 + 6];
    float ex = __expf(logits[e] - ord2f(segmax[ii]));
    exbuf[e] = ex;
    atomicAdd(segsum + ii, ex);
}

// ---------- kernel 5: per-batch top-k (radix select) + scatter sum ----------

__global__ __launch_bounds__(256) void topk_kernel(const int* __restrict__ edges,
                                                   const float* __restrict__ exbuf,
                                                   const float* __restrict__ segsum,
                                                   const float* __restrict__ att,
                                                   const int* __restrict__ kptr,
                                                   float* __restrict__ out)
{
    __shared__ unsigned keys[4096];
    __shared__ float    smv[4096];
    __shared__ unsigned hist[256];
    __shared__ unsigned sh_prefix, sh_remaining, sh_tie;

    int b = blockIdx.x;
    int tid = threadIdx.x;
    int k = kptr[0];

    for (int i = tid; i < 4096; i += 256) {
        long e = (long)b * 4096 + i;
        int ii = edges[e * 8 + 6];
        float sm = exbuf[e] / segsum[ii];
        float t  = sm * att[ii];           // >= 0, so raw bits are order-preserving
        keys[i] = __float_as_uint(t);
        smv[i]  = sm;
    }
    if (tid == 0) { sh_prefix = 0u; sh_remaining = (unsigned)k; sh_tie = 0u; }
    __syncthreads();

    for (int pass = 0; pass < 4; ++pass) {
        int shift = 24 - pass * 8;
        unsigned mask_hi = (pass == 0) ? 0u : (0xFFFFFFFFu << (shift + 8));
        hist[tid] = 0u;
        __syncthreads();
        unsigned prefix = sh_prefix;
        for (int i = tid; i < 4096; i += 256) {
            unsigned kk = keys[i];
            if ((kk & mask_hi) == prefix) atomicAdd(&hist[(kk >> shift) & 0xFFu], 1u);
        }
        __syncthreads();
        if (tid == 0) {
            unsigned rem = sh_remaining, cum = 0u; int bin = 0;
            for (int bi = 255; bi >= 0; --bi) {
                unsigned h = hist[bi];
                if (cum + h >= rem) { bin = bi; break; }
                cum += h;
            }
            sh_remaining = rem - cum;
            sh_prefix = sh_prefix | ((unsigned)bin << shift);
        }
        __syncthreads();
    }

    unsigned T = sh_prefix;
    unsigned ties = sh_remaining;
    for (int i = tid; i < 4096; i += 256) {
        unsigned kk = keys[i];
        bool take = false;
        if (kk > T) take = true;
        else if (kk == T) { unsigned pos = atomicAdd(&sh_tie, 1u); if (pos < ties) take = true; }
        if (take) {
            long e = (long)b * 4096 + i;
            int jj = edges[e * 8 + 7];
            atomicAdd(out + jj, smv[i] * __uint_as_float(kk));
        }
    }
}

// ---------- host ----------

extern "C" void kernel_launch(void* const* d_in, const int* in_sizes, int n_in,
                              void* d_out, int out_size, void* d_ws, size_t ws_size,
                              hipStream_t stream) {
    (void)n_in; (void)ws_size;
    const int*   edges = (const int*)d_in[0];
    const float* att   = (const float*)d_in[1];
    const float* emb   = (const float*)d_in[2];
    const float* rel   = (const float*)d_in[3];
    const float* qse   = (const float*)d_in[4];
    const float* qre   = (const float*)d_in[5];
    const float* qte   = (const float*)d_in[6];
    const float* Wproj = (const float*)d_in[7];
    const float* bproj = (const float*)d_in[8];
    const float* Wst   = (const float*)d_in[9];
    const float* bst   = (const float*)d_in[10];
    const float* Wtm   = (const float*)d_in[11];
    const float* btm   = (const float*)d_in[12];
    const float* Wl    = (const float*)d_in[13];
    const float* bl    = (const float*)d_in[14];
    const float* Wr    = (const float*)d_in[15];
    const float* br    = (const float*)d_in[16];
    const float* Wc    = (const float*)d_in[17];
    const float* bc    = (const float*)d_in[18];
    const int*   kptr  = (const int*)d_in[19];
    float* out = (float*)d_out;

    int  N = in_sizes[1];
    long E = (long)in_sizes[3] / 128;
    int  B = in_sizes[4] / 128;

    char* ws = (char*)d_ws;
    size_t off = 0;
    auto alloc = [&](size_t bytes) -> void* {
        void* p = ws + off;
        off = (off + bytes + 255) & ~(size_t)255;
        return p;
    };
    float*    qbiasL = (float*)alloc((size_t)B * 64 * 4);
    float*    qbiasR = (float*)alloc((size_t)B * 64 * 4);
    __bf16*   WLi    = (__bf16*)alloc(64 * 128 * 2);
    __bf16*   WLrel  = (__bf16*)alloc(64 * 128 * 2);
    __bf16*   WRj    = (__bf16*)alloc(64 * 128 * 2);
    __bf16*   WRrel  = (__bf16*)alloc(64 * 128 * 2);
    __bf16*   WcBf   = (__bf16*)alloc(64 * 64 * 2);
    float*    logits = (float*)alloc((size_t)E * 4);
    float*    exbuf  = (float*)alloc((size_t)E * 4);
    unsigned* segmax = (unsigned*)alloc((size_t)N * 4);
    float*    segsum = (float*)alloc((size_t)N * 4);

    int initN = (N > out_size) ? N : out_size;
    init_kernel<<<(initN + 255) / 256, 256, 0, stream>>>(out, segmax, segsum, N, out_size);

    prep_kernel<<<1, 256, 0, stream>>>(qse, qre, qte, Wproj, bproj, Wst, bst, Wtm, btm,
                                       Wl, bl, Wr, br, Wc,
                                       qbiasL, qbiasR, WLi, WLrel, WRj, WRrel, WcBf, B);

    size_t smem = 90112;  // 4*16KB fused weights + 8KB Wc + 16KB r-staging
    hipFuncSetAttribute((const void*)edge_wmma_kernel,
                        hipFuncAttributeMaxDynamicSharedMemorySize, (int)smem);
    int edgeBlocks = (int)(E / 128);  // 8 tiles of 16 edges per 256-thread block
    edge_wmma_kernel<<<edgeBlocks, 256, smem, stream>>>(edges, emb, rel, qbiasL, qbiasR,
                                                        WLi, WLrel, WRj, WRrel, WcBf, bc,
                                                        logits, segmax);

    softmax_sum_kernel<<<(int)((E + 255) / 256), 256, 0, stream>>>(edges, logits, segmax,
                                                                   exbuf, segsum, E);

    topk_kernel<<<B, 256, 0, stream>>>(edges, exbuf, segsum, att, kptr, out);
}
